// FlowAttention_65223373357610
// MI455X (gfx1250) — compile-verified, matrix operand order checked by reading, not run on hardware
//
#include <hip/hip_runtime.h>
#include <hip/hip_bf16.h>

// ---------------------------------------------------------------------------
// FlowAttention for MI455X (gfx1250, wave32, WMMA).
// Heavy compute = 4 big GEMMs -> v_wmma_f32_16x16x32_bf16 mixed precision.
// LDS tiles stored pre-swizzled in WMMA fragment order so each lane's
// fragment is one contiguous 32B read (2x ds_load_b128).
// ---------------------------------------------------------------------------

typedef __attribute__((ext_vector_type(16))) __bf16 v16bf;
typedef __attribute__((ext_vector_type(8)))  float  v8f;

#define BDIM 4
#define SDIM 4096
#define DDIM 1024
#define HDIM 16
#define DK   64

// fp32 -> bf16 (round-to-nearest-even) via bit manipulation
__device__ __forceinline__ unsigned short f2bf(float f) {
    union { float f; unsigned u; } x; x.f = f;
    unsigned r = x.u + 0x7FFFu + ((x.u >> 16) & 1u);
    return (unsigned short)(r >> 16);
}

// fast sigmoid: hardware v_exp_f32 + v_rcp_f32
__device__ __forceinline__ float sigmoidf_(float x) {
    return __builtin_amdgcn_rcpf(1.0f + __expf(-x));
}

// ---------------------------------------------------------------------------
// WMMA GEMM:  C[M,N] = act( A[M,K] @ B[K,N] )
// Block tile 128x64, BK=32. 256 threads = 8 waves, wave tile 32x32 (2x2 WMMA).
// Dims must be exact multiples of the tile (true for all calls here).
// LDS swizzle: value at k stored at slot [g*16 + e], g=(k>>3)&1,
//              e=(k&7)+8*(k>>4)   (inverse of the ISA 16x32 bf16 fragment map)
// ---------------------------------------------------------------------------
#define GBM 128
#define GBN 64
#define GBK 32

__global__ __launch_bounds__(256)
void flowattn_wmma_gemm(const float* __restrict__ A, const float* __restrict__ B,
                        float* __restrict__ C, int M, int N, int K,
                        int act_sigmoid, int layout_mode, int S) {
    __shared__ __attribute__((aligned(32))) __bf16 As_sw[GBM][GBK]; // [row][g*16+e]
    __shared__ __attribute__((aligned(32))) __bf16 Bs_sw[GBN][GBK]; // [col][g*16+e]

    const int tid  = threadIdx.x;
    const int lane = tid & 31;
    const int wv   = tid >> 5;          // 0..7
    const int wM   = wv & 3;            // 4 wave rows  (32 rows each)
    const int wN   = wv >> 2;           // 2 wave cols  (32 cols each)
    const int g    = lane >> 4;         // lane half
    const int ln16 = lane & 15;

    const int rowBase = blockIdx.y * GBM;
    const int colBase = blockIdx.x * GBN;

    v8f acc[2][2] = {};

    for (int k0 = 0; k0 < K; k0 += GBK) {
        // ---- A tile: 128x32 floats = 1024 float4, 4 per thread ----
        #pragma unroll
        for (int i = 0; i < 4; ++i) {
            int idx = tid + 256 * i;          // float4 index
            int r   = idx >> 3;               // 8 float4 per row
            int c4  = idx & 7;
            int k   = c4 * 4;
            const float4 a4 = *(const float4*)&A[(size_t)(rowBase + r) * K + k0 + k];
            int gs = (k >> 3) & 1;
            int e0 = (k & 7) + 8 * (k >> 4);
            uint2 p;
            p.x = (unsigned)f2bf(a4.x) | ((unsigned)f2bf(a4.y) << 16);
            p.y = (unsigned)f2bf(a4.z) | ((unsigned)f2bf(a4.w) << 16);
            *(uint2*)((void*)&As_sw[r][gs * 16 + e0]) = p;
        }
        // ---- B tile: 32x64 floats = 512 float4, 2 per thread (transposing) --
        #pragma unroll
        for (int i = 0; i < 2; ++i) {
            int idx = tid + 256 * i;
            int r   = idx >> 4;               // 16 float4 per row (64 cols)
            int c4  = idx & 15;
            const float4 b4 = *(const float4*)&B[(size_t)(k0 + r) * N + colBase + c4 * 4];
            int gs = (r >> 3) & 1;
            int e  = (r & 7) + 8 * (r >> 4);
            union { unsigned short u; __bf16 b; } cv;
            cv.u = f2bf(b4.x); Bs_sw[c4 * 4 + 0][gs * 16 + e] = cv.b;
            cv.u = f2bf(b4.y); Bs_sw[c4 * 4 + 1][gs * 16 + e] = cv.b;
            cv.u = f2bf(b4.z); Bs_sw[c4 * 4 + 2][gs * 16 + e] = cv.b;
            cv.u = f2bf(b4.w); Bs_sw[c4 * 4 + 3][gs * 16 + e] = cv.b;
        }
        // prefetch next K-slab of A into L2 (gfx1250 global_prefetch path)
        __builtin_prefetch(&A[(size_t)(rowBase + (tid >> 1)) * K + k0 + GBK], 0, 1);
        __syncthreads();

        // ---- fragments are contiguous 32B runs now ----
        v16bf afrag[2], bfrag[2];
        #pragma unroll
        for (int tm = 0; tm < 2; ++tm) {
            int arow = wM * 32 + tm * 16 + ln16;
            afrag[tm] = *(const v16bf*)((const void*)&As_sw[arow][g * 16]);
        }
        #pragma unroll
        for (int tn = 0; tn < 2; ++tn) {
            int bcol = wN * 32 + tn * 16 + ln16;
            bfrag[tn] = *(const v16bf*)((const void*)&Bs_sw[bcol][g * 16]);
        }
        #pragma unroll
        for (int tm = 0; tm < 2; ++tm)
            #pragma unroll
            for (int tn = 0; tn < 2; ++tn)
                acc[tm][tn] = __builtin_amdgcn_wmma_f32_16x16x32_bf16(
                    false, afrag[tm], false, bfrag[tn],
                    (short)0, acc[tm][tn], false, false);
        __syncthreads();
    }

    // ---- epilogue ----
    #pragma unroll
    for (int tm = 0; tm < 2; ++tm) {
        #pragma unroll
        for (int tn = 0; tn < 2; ++tn) {
            #pragma unroll
            for (int r = 0; r < 8; ++r) {
                int grow = rowBase + wM * 32 + tm * 16 + r + 8 * g;
                int gcol = colBase + wN * 32 + tn * 16 + ln16;
                float v = acc[tm][tn][r];
                if (act_sigmoid) v = sigmoidf_(v);
                if (layout_mode == 1) {
                    int h = gcol >> 6, d = gcol & 63;
                    int b = grow / S, s = grow - b * S;
                    C[(((size_t)(b * HDIM + h) * S) + s) * DK + d] = v;
                } else {
                    C[(size_t)grow * N + gcol] = v;
                }
            }
        }
    }
}

// ---------------------------------------------------------------------------
// Per-(b,h) flow reductions + softmax + kv.  One workgroup per (b,h).
// Q,K,V in [B,H,S,64] layout (Q,K already sigmoided by GEMM epilogue).
// ---------------------------------------------------------------------------
__device__ __forceinline__ float wave_sum(float v) {
    #pragma unroll
    for (int off = 16; off > 0; off >>= 1) v += __shfl_xor(v, off, 32);
    return v;
}

__global__ __launch_bounds__(256)
void flowattn_reduce(const float* __restrict__ Q, const float* __restrict__ K,
                     const float* __restrict__ V,
                     float* __restrict__ siArr, float* __restrict__ soArr,
                     float* __restrict__ csinkArr, float* __restrict__ csrcArr,
                     float* __restrict__ wArr, float* __restrict__ kvOut) {
    __shared__ float sKsum[DK], sQsum[DK], sKSO[DK], sQSI[DK];
    __shared__ float sWred[8];
    __shared__ float sScal[2];
    __shared__ float sKst[4][DK];
    __shared__ float sVst[4][DK];

    const int bh   = blockIdx.x;                 // 0..63
    const int tid  = threadIdx.x;
    const int lane = tid & 31;
    const int wv   = tid >> 5;
    const size_t base = (size_t)bh * SDIM;
    const float* Qh = Q + base * DK;
    const float* Kh = K + base * DK;
    const float* Vh = V + base * DK;

    if (tid < DK) { sKsum[tid] = 0.f; sQsum[tid] = 0.f; sKSO[tid] = 0.f; sQSI[tid] = 0.f; }
    if (tid == 0) { sScal[1] = 0.f; }
    __syncthreads();

    // ---- Phase A: K_sum, Q_sum over S ----
    {
        int d = tid & 63, rg = tid >> 6;
        float ka = 0.f, qa = 0.f;
        for (int s = rg; s < SDIM; s += 4) {
            ka += Kh[(size_t)s * DK + d];
            qa += Qh[(size_t)s * DK + d];
        }
        atomicAdd(&sKsum[d], ka);
        atomicAdd(&sQsum[d], qa);
    }
    __syncthreads();

    // ---- Phase B: sink_incoming, source_outgoing; accumulate KSO, QSI ----
    for (int s = wv; s < SDIM; s += 8) {
        float q0 = Qh[(size_t)s * DK + lane];
        float q1 = Qh[(size_t)s * DK + lane + 32];
        float k0 = Kh[(size_t)s * DK + lane];
        float k1 = Kh[(size_t)s * DK + lane + 32];
        float si = 1.0f / wave_sum(q0 * sKsum[lane] + q1 * sKsum[lane + 32]);
        float so = 1.0f / wave_sum(k0 * sQsum[lane] + k1 * sQsum[lane + 32]);
        atomicAdd(&sKSO[lane],      k0 * so);
        atomicAdd(&sKSO[lane + 32], k1 * so);
        atomicAdd(&sQSI[lane],      q0 * si);
        atomicAdd(&sQSI[lane + 32], q1 * si);
        if (lane == 0) { siArr[base + s] = si; soArr[base + s] = so; }
    }
    __syncthreads();

    // ---- Phase C: conserved_sink / conserved_source (+ running max) ----
    float wmax = -3.0e38f;
    for (int s = wv; s < SDIM; s += 8) {
        float q0 = Qh[(size_t)s * DK + lane];
        float q1 = Qh[(size_t)s * DK + lane + 32];
        float k0 = Kh[(size_t)s * DK + lane];
        float k1 = Kh[(size_t)s * DK + lane + 32];
        float cs   = wave_sum(q0 * sKSO[lane] + q1 * sKSO[lane + 32]);
        float csrc = wave_sum(k0 * sQSI[lane] + k1 * sQSI[lane + 32]);
        wmax = fmaxf(wmax, csrc);
        if (lane == 0) { csinkArr[base + s] = cs; csrcArr[base + s] = csrc; }
    }
    if (lane == 0) sWred[wv] = wmax;
    __threadfence_block();
    __syncthreads();
    if (tid == 0) {
        float m = sWred[0];
        #pragma unroll
        for (int i = 1; i < 8; ++i) m = fmaxf(m, sWred[i]);
        sScal[0] = m;
    }
    __syncthreads();

    // ---- Phase D: softmax over S of conserved_source (fast exp: arg <= 0) --
    {
        float m = sScal[0];
        float acc = 0.f;
        for (int s = tid; s < SDIM; s += 256) acc += __expf(csrcArr[base + s] - m);
        atomicAdd(&sScal[1], acc);
    }
    __syncthreads();
    {
        float m = sScal[0];
        float inv = __builtin_amdgcn_rcpf(sScal[1]);
        for (int s = tid; s < SDIM; s += 256)
            wArr[base + s] = __expf(csrcArr[base + s] - m) * inv;
    }
    __threadfence_block();
    __syncthreads();

    // ---- Phase E: kv[kd][v] = sum_s K[s][kd] * V[s][v] * w[s] ----
    float kvacc[16];
    #pragma unroll
    for (int j = 0; j < 16; ++j) kvacc[j] = 0.f;
    const int kd = tid & 63;
    const int vg = tid >> 6;  // 0..3  -> v in [vg*16, vg*16+16)
    for (int s0 = 0; s0 < SDIM; s0 += 4) {
        int r = tid >> 6, d = tid & 63;
        sKst[r][d] = Kh[(size_t)(s0 + r) * DK + d];
        sVst[r][d] = Vh[(size_t)(s0 + r) * DK + d] * wArr[base + s0 + r];
        __syncthreads();
        #pragma unroll
        for (int r2 = 0; r2 < 4; ++r2) {
            float kval = sKst[r2][kd];
            #pragma unroll
            for (int j = 0; j < 16; ++j)
                kvacc[j] += kval * sVst[r2][vg * 16 + j];
        }
        __syncthreads();
    }
    #pragma unroll
    for (int j = 0; j < 16; ++j)
        kvOut[(size_t)bh * (DK * DK) + kd * DK + vg * 16 + j] = kvacc[j];
}

// ---------------------------------------------------------------------------
// context[b,s,h*64+v] = sigmoid(csink) * sum_k (Q[s,k]*si[s]) * kv[k,v]
// grid: (B*H, S/128); kv tile cached in LDS.
// ---------------------------------------------------------------------------
__global__ __launch_bounds__(256)
void flowattn_context(const float* __restrict__ Q, const float* __restrict__ kv,
                      const float* __restrict__ siArr, const float* __restrict__ csinkArr,
                      float* __restrict__ ctx) {
    __shared__ float skv[DK][DK + 1];
    __shared__ float sQ[4][DK];

    const int bh  = blockIdx.x;
    const int b   = bh >> 4;
    const int h   = bh & 15;
    const int s0  = blockIdx.y * 128;
    const int tid = threadIdx.x;
    const size_t base = (size_t)bh * SDIM;
    const float* Qh = Q + base * DK;

    #pragma unroll
    for (int i = 0; i < 16; ++i) {
        int idx = tid + 256 * i;
        skv[idx >> 6][idx & 63] = kv[(size_t)bh * (DK * DK) + idx];
    }
    __syncthreads();

    const int v  = tid & 63;
    const int rg = tid >> 6;
    for (int it = 0; it < 32; ++it) {
        int srow = s0 + it * 4;
        {
            int r = tid >> 6, d = tid & 63;
            sQ[r][d] = Qh[(size_t)(srow + r) * DK + d] * siArr[base + srow + r];
        }
        __syncthreads();
        float acc = 0.f;
        #pragma unroll
        for (int k = 0; k < DK; ++k) acc += sQ[rg][k] * skv[k][v];
        int s = srow + rg;
        float gate = sigmoidf_(csinkArr[base + s]);
        ctx[((size_t)(b * SDIM + s) * DDIM) + h * DK + v] = acc * gate;
        __syncthreads();
    }
}

// ---------------------------------------------------------------------------
// out = LayerNorm(fc_out + residual) * gamma + beta ; one block per row.
// ---------------------------------------------------------------------------
__global__ __launch_bounds__(256)
void flowattn_ln(const float* __restrict__ fcout, const float* __restrict__ resid,
                 const float* __restrict__ gamma, const float* __restrict__ beta,
                 float* __restrict__ out) {
    __shared__ float red[256];
    const int row = blockIdx.x;
    const int tid = threadIdx.x;
    const size_t rb = (size_t)row * DDIM;

    float x[4];
    float lsum = 0.f;
    #pragma unroll
    for (int j = 0; j < 4; ++j) {
        int i = tid + j * 256;
        x[j] = fcout[rb + i] + resid[rb + i];
        lsum += x[j];
    }
    red[tid] = lsum; __syncthreads();
    for (int st = 128; st > 0; st >>= 1) {
        if (tid < st) red[tid] += red[tid + st];
        __syncthreads();
    }
    float mu = red[0] * (1.0f / DDIM);
    __syncthreads();

    float lsq = 0.f;
    #pragma unroll
    for (int j = 0; j < 4; ++j) { float d = x[j] - mu; lsq += d * d; }
    red[tid] = lsq; __syncthreads();
    for (int st = 128; st > 0; st >>= 1) {
        if (tid < st) red[tid] += red[tid + st];
        __syncthreads();
    }
    float rstd = rsqrtf(red[0] * (1.0f / DDIM) + 1e-5f);

    #pragma unroll
    for (int j = 0; j < 4; ++j) {
        int i = tid + j * 256;
        out[rb + i] = (x[j] - mu) * rstd * gamma[i] + beta[i];
    }
}

// ---------------------------------------------------------------------------
extern "C" void kernel_launch(void* const* d_in, const int* in_sizes, int n_in,
                              void* d_out, int out_size, void* d_ws, size_t ws_size,
                              hipStream_t stream) {
    const float* input_Q = (const float*)d_in[0];
    const float* input_K = (const float*)d_in[1];
    const float* input_V = (const float*)d_in[2];
    const float* W_Q     = (const float*)d_in[3];
    const float* W_K     = (const float*)d_in[4];
    const float* W_V     = (const float*)d_in[5];
    const float* W_fc    = (const float*)d_in[6];
    const float* ln_g    = (const float*)d_in[7];
    const float* ln_b    = (const float*)d_in[8];
    float* out = (float*)d_out;

    const int M = BDIM * SDIM;      // 16384
    const int N = DDIM;             // 1024
    const int Kd = DDIM;            // 1024

    // workspace layout (floats); context aliases V slot, fcout aliases K slot
    float* ws = (float*)d_ws;
    const size_t BIG = (size_t)BDIM * HDIM * SDIM * DK;   // 16,777,216
    const size_t SMALL = (size_t)BDIM * HDIM * SDIM;      //    262,144
    float* Qp    = ws;
    float* Kp    = ws + BIG;
    float* Vp    = ws + 2 * BIG;
    float* siA   = ws + 3 * BIG;
    float* soA   = siA + SMALL;
    float* cskA  = soA + SMALL;
    float* csrA  = cskA + SMALL;
    float* wA    = csrA + SMALL;
    float* kvA   = wA + SMALL;                            // B*H*64*64 = 262,144
    float* ctx   = Vp;    // alias: V not needed after flowattn_reduce
    float* fcout = Kp;    // alias: K not needed after flowattn_reduce

    dim3 gemmGrid(N / GBN, M / GBM);   // (16, 128)
    dim3 blk(256);

    // 1-3: projections, sigmoid fused on Q/K, head-split output layout
    flowattn_wmma_gemm<<<gemmGrid, blk, 0, stream>>>(input_Q, W_Q, Qp, M, N, Kd, 1, 1, SDIM);
    flowattn_wmma_gemm<<<gemmGrid, blk, 0, stream>>>(input_K, W_K, Kp, M, N, Kd, 1, 1, SDIM);
    flowattn_wmma_gemm<<<gemmGrid, blk, 0, stream>>>(input_V, W_V, Vp, M, N, Kd, 0, 1, SDIM);

    // 4: flow conservation reductions + softmax + kv per (b,h)
    flowattn_reduce<<<dim3(BDIM * HDIM), blk, 0, stream>>>(Qp, Kp, Vp, siA, soA, cskA, csrA, wA, kvA);

    // 5: context
    flowattn_context<<<dim3(BDIM * HDIM, SDIM / 128), blk, 0, stream>>>(Qp, kvA, siA, cskA, ctx);

    // 6: output projection
    flowattn_wmma_gemm<<<gemmGrid, blk, 0, stream>>>(ctx, W_fc, fcout, M, N, Kd, 0, 0, SDIM);

    // 7: residual + LayerNorm
    flowattn_ln<<<dim3(M), blk, 0, stream>>>(fcout, input_Q, ln_g, ln_b, out);
}
